// fine_LSTM_85023172592224
// MI455X (gfx1250) — matrix operand order repaired and checked by reading, not compiled, not run on hardware
//
#include <hip/hip_runtime.h>
#include <math.h>

#define ITER 8
#define L 4096
#define D 512
#define H 256

typedef __bf16 bf16_t;
typedef __bf16 v16bf __attribute__((ext_vector_type(16)));
typedef __bf16 v8bf  __attribute__((ext_vector_type(8)));
typedef float  v8f   __attribute__((ext_vector_type(8)));

#define A_STRIDE 520   // bf16 elements per LDS row (512 + 8 pad, keeps 16B alignment)

// ---------------------------------------------------------------------------
// Ws (D x H, f32, row-major) -> Wst (H x D, bf16) : transpose + downconvert
// so the WMMA B-operand fragment (two runs of 8 contiguous K values per lane) is
// two 16-byte global loads.
// ---------------------------------------------------------------------------
__global__ __launch_bounds__(256) void ws_convert_kernel(const float* __restrict__ Ws,
                                                         bf16_t* __restrict__ Wst) {
  int i = blockIdx.x * 256 + threadIdx.x;   // i in [0, D*H)
  int k = i >> 8;          // row in D   (H == 256)
  int n = i & (H - 1);     // col in H
  Wst[(size_t)n * D + k] = (bf16_t)Ws[i];
}

// ---------------------------------------------------------------------------
// gate_head: out[row] = sum_n( Wh[row%L, n] * tanh( (X @ Ws)[row,n] + bs[n] ) ) + bh[row%L]
// One workgroup = 16 rows x 256 cols. 8 waves, each wave owns two 16x16
// N-tiles. K loop: V_WMMA_F32_16X16X32_BF16, A staged in LDS as bf16.
// ---------------------------------------------------------------------------
__global__ __launch_bounds__(256) void gate_head_kernel(
    const float*  __restrict__ X,     // nrows x D, f32
    const bf16_t* __restrict__ Wst,   // H x D, bf16 (transposed Ws)
    const float*  __restrict__ bs,    // H
    const float*  __restrict__ Wh,    // L x H
    const float*  __restrict__ bh,    // L
    float*        __restrict__ out)   // nrows
{
  __shared__ __attribute__((aligned(16))) bf16_t ldsA[16 * A_STRIDE];
  __shared__ float red[16];

  const int tid = threadIdx.x;
  const int m0  = blockIdx.x * 16;

  // Stage + convert A block (16 x 512 f32 -> bf16) into LDS, coalesced.
  for (int i = tid; i < 16 * D; i += 256) {
    int r = i >> 9, col = i & (D - 1);
    ldsA[r * A_STRIDE + col] = (bf16_t)X[(size_t)(m0 + r) * D + col];
  }
  if (tid < 16) red[tid] = 0.0f;
  __syncthreads();

  const int wave = tid >> 5;
  const int lane = tid & 31;
  const int grp  = lane >> 4;   // lane half: selects K sub-blocks per ISA layout
  const int r    = lane & 15;
  const int n0a  = wave * 32;
  const int n0b  = n0a + 16;

  v8f acc0 = {0.f, 0.f, 0.f, 0.f, 0.f, 0.f, 0.f, 0.f};
  v8f acc1 = acc0;

  // 16-bit A 16x32 layout: element e of the lane's fragment holds
  // K = kb + (e>>3)*16 + grp*8 + (e&7), row M = r.
  const bf16_t* arow  = &ldsA[r * A_STRIDE + grp * 8];
  // B 32x16 layout (same K order as A): element e holds
  // K = kb + (e>>3)*16 + grp*8 + (e&7), N = n0 + r (rows of the transposed Wst).
  const bf16_t* b0row = &Wst[(size_t)(n0a + r) * D + grp * 8];
  const bf16_t* b1row = &Wst[(size_t)(n0b + r) * D + grp * 8];

  for (int kb = 0; kb < D; kb += 32) {
    v8bf a_lo = *(const v8bf*)(arow + kb);        // K = kb+grp*8 .. +7
    v8bf a_hi = *(const v8bf*)(arow + kb + 16);   // K = kb+16+grp*8 .. +7
    v16bf a;
#pragma unroll
    for (int e = 0; e < 8; ++e) { a[e] = a_lo[e]; a[8 + e] = a_hi[e]; }
    v8bf b0_lo = *(const v8bf*)(b0row + kb);
    v8bf b0_hi = *(const v8bf*)(b0row + kb + 16);
    v8bf b1_lo = *(const v8bf*)(b1row + kb);
    v8bf b1_hi = *(const v8bf*)(b1row + kb + 16);
    v16bf b0, b1;
#pragma unroll
    for (int e = 0; e < 8; ++e) { b0[e] = b0_lo[e]; b0[8 + e] = b0_hi[e]; b1[e] = b1_lo[e]; b1[8 + e] = b1_hi[e]; }
    acc0 = __builtin_amdgcn_wmma_f32_16x16x32_bf16(false, a, false, b0,
                                                   (short)0, acc0, false, false);
    acc1 = __builtin_amdgcn_wmma_f32_16x16x32_bf16(false, a, false, b1,
                                                   (short)0, acc1, false, false);
  }

  // Epilogue fused: s = tanh(acc + bs); partial = s * Wh; reduce over N via LDS.
  // C/D layout: VGPR i -> M = i (lanes 0-15) / 8+i (lanes 16-31), N = n0 + r.
#pragma unroll
  for (int i = 0; i < 8; ++i) {
    const int Mi = i + 8 * grp;
    const int lM = (m0 + Mi) & (L - 1);   // Wh/bh index wraps per L rows
    {
      const int N = n0a + r;
      float s = tanhf(acc0[i] + bs[N]);
      atomicAdd(&red[Mi], s * Wh[(size_t)lM * H + N]);
    }
    {
      const int N = n0b + r;
      float s = tanhf(acc1[i] + bs[N]);
      atomicAdd(&red[Mi], s * Wh[(size_t)lM * H + N]);
    }
  }
  __syncthreads();
  if (tid < 16) {
    int grow = m0 + tid;
    out[grow] = red[tid] + bh[grow & (L - 1)];
  }
}

// ---------------------------------------------------------------------------
__device__ __forceinline__ float wave_sum(float v) {
#pragma unroll
  for (int off = 16; off > 0; off >>= 1) v += __shfl_down(v, off, 32);
  return v;
}

// init: h = emb0, c = rois0, out[0] = dec(emb0)*size + rois0
__global__ __launch_bounds__(256) void init_kernel(
    const float* __restrict__ emb0, const float* __restrict__ rois,
    const float* __restrict__ Wx, const float* __restrict__ bx,
    const float* __restrict__ Wy, const float* __restrict__ by,
    const float* __restrict__ Wz, const float* __restrict__ bz,
    float* __restrict__ h, float* __restrict__ c, float* __restrict__ out0)
{
  const int row  = blockIdx.x * 8 + (threadIdx.x >> 5);
  const int lane = threadIdx.x & 31;
  const size_t base = (size_t)row * D;
  float px = 0.f, py = 0.f, pz = 0.f;
  for (int j = lane; j < D; j += 32) {
    float e = emb0[base + j];
    h[base + j] = e;
    px += Wx[base + j] * e;
    py += Wy[base + j] * e;
    pz += Wz[base + j] * e;
  }
  px = wave_sum(px); py = wave_sum(py); pz = wave_sum(pz);
  if (lane == 0) {
    float r0 = rois[row * 3 + 0], r1 = rois[row * 3 + 1], r2 = rois[row * 3 + 2];
    c[row * 3 + 0] = r0; c[row * 3 + 1] = r1; c[row * 3 + 2] = r2;
    out0[row * 3 + 0] = (px + bx[row]) * (1.0f / 767.0f) + r0;
    out0[row * 3 + 1] = (py + by[row]) * (1.0f / 767.0f) + r1;
    out0[row * 3 + 2] = (pz + bz[row]) * (1.0f / 575.0f) + r2;
  }
}

// step: g = softmax([gf, ga]); h = h*g0 + emb*g1; c = c*g0 + prev_pred*g1;
//       out[t] = dec(h)*size + c
__global__ __launch_bounds__(256) void step_kernel(
    const float* __restrict__ embt, const float* __restrict__ gf,
    const float* __restrict__ gat,  const float* __restrict__ prev_pred,
    const float* __restrict__ Wx, const float* __restrict__ bx,
    const float* __restrict__ Wy, const float* __restrict__ by,
    const float* __restrict__ Wz, const float* __restrict__ bz,
    float* __restrict__ h, float* __restrict__ c, float* __restrict__ outt)
{
  const int row  = blockIdx.x * 8 + (threadIdx.x >> 5);
  const int lane = threadIdx.x & 31;
  const float f = gf[row], a = gat[row];
  const float m = fmaxf(f, a);
  const float e0 = __expf(f - m), e1 = __expf(a - m);
  const float inv = 1.0f / (e0 + e1);
  const float g0 = e0 * inv, g1 = e1 * inv;

  const size_t base = (size_t)row * D;
  float px = 0.f, py = 0.f, pz = 0.f;
  for (int j = lane; j < D; j += 32) {
    float hv = h[base + j] * g0 + embt[base + j] * g1;
    h[base + j] = hv;
    px += Wx[base + j] * hv;
    py += Wy[base + j] * hv;
    pz += Wz[base + j] * hv;
  }
  px = wave_sum(px); py = wave_sum(py); pz = wave_sum(pz);
  if (lane == 0) {
    float c0 = c[row * 3 + 0] * g0 + prev_pred[row * 3 + 0] * g1;
    float c1 = c[row * 3 + 1] * g0 + prev_pred[row * 3 + 1] * g1;
    float c2 = c[row * 3 + 2] * g0 + prev_pred[row * 3 + 2] * g1;
    c[row * 3 + 0] = c0; c[row * 3 + 1] = c1; c[row * 3 + 2] = c2;
    outt[row * 3 + 0] = (px + bx[row]) * (1.0f / 767.0f) + c0;
    outt[row * 3 + 1] = (py + by[row]) * (1.0f / 767.0f) + c1;
    outt[row * 3 + 2] = (pz + bz[row]) * (1.0f / 575.0f) + c2;
  }
}

// ---------------------------------------------------------------------------
extern "C" void kernel_launch(void* const* d_in, const int* in_sizes, int n_in,
                              void* d_out, int out_size, void* d_ws, size_t ws_size,
                              hipStream_t stream) {
  (void)in_sizes; (void)n_in; (void)out_size; (void)ws_size;

  const float* emb  = (const float*)d_in[0];   // (8, L, D)
  const float* rois = (const float*)d_in[1];   // (1, L, 3)
  const float* Wx   = (const float*)d_in[2];
  const float* bx   = (const float*)d_in[3];
  const float* Wy   = (const float*)d_in[4];
  const float* by   = (const float*)d_in[5];
  const float* Wz   = (const float*)d_in[6];
  const float* bz   = (const float*)d_in[7];
  const float* Ws   = (const float*)d_in[8];   // (D, H)
  const float* bs   = (const float*)d_in[9];
  const float* Wh   = (const float*)d_in[10];  // (L, H)
  const float* bh   = (const float*)d_in[11];
  float* out = (float*)d_out;                  // (8, L, 3)

  // Workspace carve-up.
  char* w = (char*)d_ws;
  float*  hst = (float*)w;              w += (size_t)L * D * sizeof(float); // 8 MB
  float*  cst = (float*)w;              w += (size_t)L * 3 * sizeof(float);
  float*  ga  = (float*)w;              w += (size_t)(ITER - 1) * L * sizeof(float);
  float*  gfb = (float*)w;              w += (size_t)L * sizeof(float);
  bf16_t* Wst = (bf16_t*)w;             // H*D bf16 = 256 KB

  // 1) Ws -> bf16 transposed.
  ws_convert_kernel<<<(D * H) / 256, 256, 0, stream>>>(Ws, Wst);

  // 2) t=0: h=emb0, c=rois0, out[0]=dec(emb0)*size+rois0.
  init_kernel<<<L / 8, 256, 0, stream>>>(emb, rois, Wx, bx, Wy, by, Wz, bz,
                                         hst, cst, out);

  // 3) ga for all future steps in one batched WMMA pass (28672 rows).
  gate_head_kernel<<<((ITER - 1) * L) / 16, 256, 0, stream>>>(
      emb + (size_t)L * D, Wst, bs, Wh, bh, ga);

  // 4) Serial recurrence: gate_head(h) -> fused gate/update/dec per step.
  for (int t = 1; t < ITER; ++t) {
    gate_head_kernel<<<L / 16, 256, 0, stream>>>(hst, Wst, bs, Wh, bh, gfb);
    step_kernel<<<L / 8, 256, 0, stream>>>(
        emb + (size_t)t * L * D, gfb, ga + (size_t)(t - 1) * L,
        out + (size_t)(t - 1) * L * 3,
        Wx, bx, Wy, by, Wz, bz,
        hst, cst, out + (size_t)t * L * 3);
  }
}
